// OuterProductMean_927712936261
// MI455X (gfx1250) — compile-verified, matrix operand order checked by reading, not run on hardware
//
#include <hip/hip_runtime.h>
#include <hip/hip_bf16.h>

#define S_DIM 512
#define N_RES 384
#define D_MSA 64
#define D_HID 32
#define D_PAIR 128
#define CH (N_RES * D_HID) /* 12288 */

typedef __bf16 v8bf  __attribute__((ext_vector_type(8)));
typedef __bf16 v16bf __attribute__((ext_vector_type(16)));
typedef float  v8f   __attribute__((ext_vector_type(8)));

#define WMMA_BF16(a, b, c) \
    __builtin_amdgcn_wmma_f32_16x16x32_bf16(false, (a), false, (b), (short)0, (c), false, false)

// ---- global->register fragment (row-major [channel, K] bf16, K contiguous) ----
// 16-bit A/B layout: lane L holds row/col = L%16, base = (L>=16)?8:0;
// VGPR0..3: K=base+0..7, VGPR4..7: K=16+base+0..7 -> two 16B loads at +0 / +16 elems.
static __device__ __forceinline__ v16bf load_frag(const __bf16* p) {
    v8bf lo = *(const v8bf*)(p);
    v8bf hi = *(const v8bf*)(p + 16);
    return __builtin_shufflevector(lo, hi, 0, 1, 2, 3, 4, 5, 6, 7,
                                           8, 9, 10, 11, 12, 13, 14, 15);
}

// ---- LDS fragment-major load: frag = [chunk0: 32 lanes x 16B][chunk1: 32 lanes x 16B]
// Lane-contiguous 16B accesses -> conflict-free ds_load_b128.
static __device__ __forceinline__ v16bf lds_frag(const __bf16* p) {
    v8bf lo = *(const v8bf*)(p);          // chunk c1=0
    v8bf hi = *(const v8bf*)(p + 256);    // chunk c1=1 (+512 bytes)
    return __builtin_shufflevector(lo, hi, 0, 1, 2, 3, 4, 5, 6, 7,
                                           8, 9, 10, 11, 12, 13, 14, 15);
}

// ---- CDNA5 async copy: 16B global -> LDS, tracked by ASYNCcnt ----
static __device__ __forceinline__ void async_copy16(unsigned lds_off, const __bf16* g) {
    asm volatile("global_load_async_to_lds_b128 %0, %1, off"
                 :: "v"(lds_off), "v"((unsigned long long)(uintptr_t)g)
                 : "memory");
}
static __device__ __forceinline__ void wait_async0() {
    asm volatile("s_wait_asynccnt 0x0" ::: "memory");
}

// ---------------- kernel 0: constants prep ----------------
__global__ void opm_setup_kernel(const float* __restrict__ w_out,
                                 const unsigned char* __restrict__ msa_mask,
                                 __bf16* __restrict__ w2t,
                                 float* __restrict__ inv_num) {
    int t = threadIdx.x;
    for (int idx = t; idx < D_PAIR * 1024; idx += 256) {
        int p = idx >> 10;
        int k = idx & 1023;
        w2t[idx] = (__bf16)w_out[k * D_PAIR + p];
    }
    __shared__ float red[256];
    float s = 0.f;
    for (int i = t; i < S_DIM; i += 256) s += msa_mask[i] ? 1.f : 0.f;
    red[t] = s;
    __syncthreads();
    for (int off = 128; off > 0; off >>= 1) {
        if (t < off) red[t] += red[t + off];
        __syncthreads();
    }
    if (t == 0) {
        float nm = red[0] < 1e-5f ? 1e-5f : red[0];
        *inv_num = 1.0f / nm;
    }
}

// ---------------- kernel 1: LN + hidden projection + bf16 transpose ----------------
__global__ void __launch_bounds__(256) opm_prep_kernel(
    const float* __restrict__ msa,
    const unsigned char* __restrict__ msa_mask,
    const float* __restrict__ gamma,
    const float* __restrict__ beta,
    const float* __restrict__ w_hidden,
    __bf16* __restrict__ At,
    __bf16* __restrict__ Bt) {

    __shared__ float Wlds[64 * 64];
    __shared__ float Xlds[8][64];

    int tid = threadIdx.x;
    for (int idx = tid; idx < 64 * 64; idx += 256) Wlds[idx] = w_hidden[idx];

    int wave = tid >> 5, lane = tid & 31;
    int task = blockIdx.x * 8 + wave;
    int s = task / N_RES;
    int i = task % N_RES;

    const float* xp = msa + (size_t)(s * N_RES + i) * D_MSA;
    float x0 = xp[2 * lane], x1 = xp[2 * lane + 1];
    float sum = x0 + x1;
    float sq = x0 * x0 + x1 * x1;
#pragma unroll
    for (int off = 16; off > 0; off >>= 1) {
        sum += __shfl_xor(sum, off);
        sq  += __shfl_xor(sq, off);
    }
    float mu   = sum * (1.f / 64.f);
    float var  = sq * (1.f / 64.f) - mu * mu;
    float rstd = rsqrtf(var + 1e-5f);
    Xlds[wave][2 * lane]     = (x0 - mu) * rstd * gamma[2 * lane]     + beta[2 * lane];
    Xlds[wave][2 * lane + 1] = (x1 - mu) * rstd * gamma[2 * lane + 1] + beta[2 * lane + 1];
    __syncthreads();

    float sm = msa_mask[s] ? 1.f : 0.f;
    float ha = 0.f, hb = 0.f;
#pragma unroll 8
    for (int d = 0; d < 64; ++d) {
        float xv = Xlds[wave][d];
        ha += xv * Wlds[d * 64 + lane];
        hb += xv * Wlds[d * 64 + 32 + lane];
    }
    size_t row = (size_t)i * 32 + lane;
    At[row * S_DIM + s] = (__bf16)(ha * sm);
    Bt[row * S_DIM + s] = (__bf16)(hb * sm);
}

// ---------------- kernel 2: fused double GEMM ----------------
// WG = 2 i's (M=64) x 16 j's (N=512), K=512. 8 waves / 256 threads.
// A panel (64x512 bf16 = 64KB) async-copied to LDS in fragment-major order.
// C (64x512 f32 regs) -> LDS bf16 (fragment-major, aliasing A region) -> phase 2.
__global__ void __launch_bounds__(256) opm_main_kernel(
    const __bf16* __restrict__ At,
    const __bf16* __restrict__ Bt,
    const __bf16* __restrict__ w2t,
    const float* __restrict__ inv_num_p,
    const unsigned char* __restrict__ mask,
    const float* __restrict__ b_out,
    float* __restrict__ out) {

    int i0 = blockIdx.y * 2;
    int j0 = blockIdx.x * 16;
    int tid = threadIdx.x;
    int wave = tid >> 5, lane = tid & 31;
    int lrow  = lane & 15;
    int khalf = (lane >> 4) * 8;

    __shared__ __align__(128) unsigned char smem[64 * 1024];
    unsigned smem_base = (unsigned)(uintptr_t)&smem[0];

    // ---- async-copy A panel into fragment-major LDS (layout transform in flight) ----
    // global chunk id: gr = row (0..63 -> channel i0*32+gr), kc = 16B chunk (0..63)
    // frag id f = kt*4+mt; dst = f*1024 + c1*512 + (lrow + 16*b)*16
    const __bf16* Apanel = At + (size_t)i0 * 32 * S_DIM;
#pragma unroll
    for (int it = 0; it < 16; ++it) {
        int id = tid + it * 256;             // 4096 chunks of 16B = 64KB
        int gr = id >> 6;
        int kc = id & 63;
        int kt = kc >> 2;
        int b  = kc & 1;
        int c1 = (kc >> 1) & 1;
        int mt = gr >> 4;
        unsigned dst = smem_base +
            (unsigned)(((kt * 4 + mt) * 1024) + c1 * 512 + (((gr & 15) + 16 * b) * 16));
        async_copy16(dst, Apanel + (size_t)gr * S_DIM + kc * 8);
    }
    wait_async0();
    __syncthreads();

    // ---- phase 1: C[64,512] += A[64,512] x B^T ----
    v8f acc[4][4];
#pragma unroll
    for (int mt = 0; mt < 4; ++mt)
#pragma unroll
        for (int nt = 0; nt < 4; ++nt) acc[mt][nt] = {};

    const __bf16* Bbase =
        Bt + ((size_t)j0 * 32 + (size_t)(wave * 4) * 16 + lrow) * S_DIM + khalf;

#pragma unroll 1
    for (int kt = 0; kt < 16; ++kt) {
        v16bf af[4];
#pragma unroll
        for (int mt = 0; mt < 4; ++mt)
            af[mt] = lds_frag((const __bf16*)(smem + (kt * 4 + mt) * 1024 + lane * 16));
#pragma unroll
        for (int nt = 0; nt < 4; ++nt) {
            const __bf16* bp = Bbase + (size_t)nt * 16 * S_DIM + kt * 32;
            v16bf bfrag = load_frag(bp);
            __builtin_prefetch(bp + 32, 0, 1);   // unconditional, speculative
#pragma unroll
            for (int mt = 0; mt < 4; ++mt)
                acc[mt][nt] = WMMA_BF16(af[mt], bfrag, acc[mt][nt]);
        }
    }

    // ---- stage C to LDS (bf16, fragment-major for phase 2), aliasing A region ----
    __syncthreads();   // all waves done reading A
#pragma unroll
    for (int mt = 0; mt < 4; ++mt)
#pragma unroll
        for (int nt = 0; nt < 4; ++nt) {
            int n = (wave * 4 + nt) * 16 + lrow;
#pragma unroll
            for (int r = 0; r < 8; ++r) {
                int m  = mt * 16 + r + ((lane >> 4) << 3);
                int il = m >> 5, d = m & 31;       // i-local, k-tile row
                int jj = n >> 5, e = n & 31;       // output row, k within tile
                unsigned addr = (unsigned)(((il * 32 + d) * 1024) + ((e >> 4) * 512) +
                                           ((jj + ((e >> 3) & 1) * 16) * 16) + ((e & 7) * 2));
                *(__bf16*)(smem + addr) = (__bf16)acc[mt][nt][r];
            }
        }
    __syncthreads();

    // ---- phase 2: out[16 j, 128 p] = Cflat[16,1024] x w2t^T, per i ----
    float inv = *inv_num_p;
    int p = wave * 16 + lrow;
    float bo = b_out[p];
    const __bf16* wp = w2t + (size_t)p * 1024 + khalf;

    v8f acc2[2] = {{}, {}};
#pragma unroll 1
    for (int kt = 0; kt < 32; ++kt) {             // K = 1024 (d*32+e), shared B frag
        v16bf b2 = load_frag(wp + kt * 32);
#pragma unroll
        for (int il = 0; il < 2; ++il) {
            v16bf a2 = lds_frag((const __bf16*)(smem + (il * 32 + kt) * 1024 + lane * 16));
            acc2[il] = WMMA_BF16(a2, b2, acc2[il]);
        }
    }

#pragma unroll
    for (int il = 0; il < 2; ++il) {
        int i = i0 + il;
        float mi = mask[i] ? inv : 0.f;
#pragma unroll
        for (int r = 0; r < 8; ++r) {
            int jj = r + ((lane >> 4) << 3);
            int j  = j0 + jj;
            float scale = mask[j] ? mi : 0.f;
            out[((size_t)i * N_RES + j) * D_PAIR + p] = acc2[il][r] * scale + bo;
        }
    }
}

extern "C" void kernel_launch(void* const* d_in, const int* in_sizes, int n_in,
                              void* d_out, int out_size, void* d_ws, size_t ws_size,
                              hipStream_t stream) {
    const float*         msa      = (const float*)d_in[0];
    const unsigned char* mask     = (const unsigned char*)d_in[1];  // bool[1,384]
    const unsigned char* msa_mask = (const unsigned char*)d_in[2];  // bool[1,512]
    const float*         gamma    = (const float*)d_in[3];
    const float*         beta     = (const float*)d_in[4];
    const float*         w_hidden = (const float*)d_in[5];
    const float*         w_out    = (const float*)d_in[6];
    const float*         b_out    = (const float*)d_in[7];
    float*               out      = (float*)d_out;

    // Workspace (~25.4 MB): At | Bt | w2t | inv_num
    __bf16* At  = (__bf16*)d_ws;
    __bf16* Bt  = At + (size_t)CH * S_DIM;
    __bf16* w2t = Bt + (size_t)CH * S_DIM;
    float*  inv_num = (float*)(w2t + (size_t)D_PAIR * 1024);

    opm_setup_kernel<<<1, 256, 0, stream>>>(w_out, msa_mask, w2t, inv_num);
    opm_prep_kernel<<<(S_DIM * N_RES) / 8, 256, 0, stream>>>(
        msa, msa_mask, gamma, beta, w_hidden, At, Bt);
    dim3 g2(N_RES / 16, N_RES / 2);
    opm_main_kernel<<<g2, 256, 0, stream>>>(At, Bt, w2t, inv_num, mask, b_out, out);
}
